// HGTAttentionHead_17076789969450
// MI455X (gfx1250) — matrix operand order repaired and checked by reading, not compiled, not run on hardware
//
#include <hip/hip_runtime.h>
#include <hip/hip_fp16.h>

typedef _Float16 f16x8  __attribute__((ext_vector_type(8)));
typedef _Float16 f16x16 __attribute__((ext_vector_type(16)));
typedef float    f32x8  __attribute__((ext_vector_type(8)));

#define HID 128
#define NTY 4
#define ETY 8

// ---------- helpers ----------

// Monotone float<->uint encoding so atomicMax(uint) implements float max.
__device__ __forceinline__ unsigned encF(float f) {
  unsigned u = __float_as_uint(f);
  return (u & 0x80000000u) ? ~u : (u | 0x80000000u);
}
__device__ __forceinline__ float decF(unsigned u) {
  return (u & 0x80000000u) ? __uint_as_float(u & 0x7fffffffu)
                           : __uint_as_float(~u);
}

// B-fragment storage: [mat][kc:4][nt:8][lane:32][i:16] halves, 16384 halves/mat.
__device__ __forceinline__ f16x16 load_bfrag(const _Float16* Wf, int mat, int kc,
                                             int nt, int lane) {
  return *(const f16x16*)(Wf + ((size_t)(((mat * 4 + kc) * 8) + nt) * 32 + lane) * 16);
}

// ---------- setup kernels ----------

__global__ void __launch_bounds__(256) k_init(float* out, unsigned* segmax,
                                              float* segsum, int N) {
  int i = blockIdx.x * 256 + threadIdx.x;
  if (i < N * HID) out[i] = 0.f;
  if (i < N) { segmax[i] = encF(-1e9f); segsum[i] = 0.f; }
}

// Convert + swizzle all 20 weight matrices into WMMA B-fragment order.
// B 32x16 f16 layout: lane<16 holds K{0..7,16..23}, lane>=16 K{8..15,24..31};
// half index i -> K = kc*32 + (i&7) + (i>>3)*16 + (lane>=16 ? 8 : 0), N = nt*16 + (lane&15).
__global__ void __launch_bounds__(256) k_pack_w(const float* WQ, const float* WK,
                                                const float* WV, const float* WE,
                                                _Float16* Wf) {
  int id = blockIdx.x * 256 + threadIdx.x;
  const int total = (3 * NTY + ETY) * HID * HID;
  if (id >= total) return;
  int i    = id & 15;
  int lane = (id >> 4) & 31;
  int nt   = (id >> 9) & 7;
  int kc   = (id >> 12) & 3;
  int mat  = id >> 14;
  const float* srcM;
  if (mat < NTY)            srcM = WQ + (size_t)mat * HID * HID;
  else if (mat < 2 * NTY)   srcM = WK + (size_t)(mat - NTY) * HID * HID;
  else if (mat < 3 * NTY)   srcM = WV + (size_t)(mat - 2 * NTY) * HID * HID;
  else                      srcM = WE + (size_t)(mat - 3 * NTY) * HID * HID;
  int k = kc * 32 + (i & 7) + ((i >> 3) << 4) + ((lane >= 16) ? 8 : 0);
  int n = nt * 16 + (lane & 15);
  Wf[id] = (_Float16)srcM[(size_t)k * HID + n];
}

// ---------- phase 1: type-masked projections Q,K,V  (WMMA) ----------
// One wave per 16-node tile. A 16x32 f16 fragment: lane<16 row=lane, K{0..7,16..23};
// lane>=16 row=lane-16, K{8..15,24..31}.
// Loop order m -> t -> kc -> nt: masked A built once per (t,kc), consumed by
// 8 consecutive WMMAs over the N tiles (B swaps, A held).

__global__ void __launch_bounds__(256) k_project(const float* x, const int* ntype,
                                                 const _Float16* Wf,
                                                 _Float16* Qh, _Float16* Kh,
                                                 float* Vf, int N) {
  int nTiles = (N + 15) >> 4;
  int tile = blockIdx.x * 8 + (threadIdx.x >> 5);
  if (tile >= nTiles) return;                 // wave-uniform
  int lane = threadIdx.x & 31;
  int row15 = lane & 15;
  int hb = lane >> 4;                          // 0 or 1
  int node = tile * 16 + row15;
  int nodeC = node < N ? node : N - 1;
  int typ = ntype[nodeC];

  f16x16 a[4];
  #pragma unroll
  for (int kc = 0; kc < 4; kc++) {
    const float* p = x + (size_t)nodeC * HID + kc * 32 + hb * 8;
    #pragma unroll
    for (int j = 0; j < 8; j++) {
      a[kc][j]     = (_Float16)p[j];
      a[kc][j + 8] = (_Float16)p[j + 16];
    }
  }

  #pragma unroll
  for (int m = 0; m < 3; m++) {                // Q, K, V
    f32x8 acc[8];
    #pragma unroll
    for (int nt = 0; nt < 8; nt++)
      #pragma unroll
      for (int v = 0; v < 8; v++) acc[nt][v] = 0.f;

    #pragma unroll
    for (int t = 0; t < NTY; t++) {
      bool keep = (typ == t);                  // per-lane row mask
      #pragma unroll
      for (int kc = 0; kc < 4; kc++) {
        f16x16 am;
        #pragma unroll
        for (int j = 0; j < 16; j++) am[j] = keep ? a[kc][j] : (_Float16)0;
        #pragma unroll
        for (int nt = 0; nt < 8; nt++) {
          f16x16 b = load_bfrag(Wf, m * NTY + t, kc, nt, lane);
          acc[nt] = __builtin_amdgcn_wmma_f32_16x16x32_f16(
              false, am, false, b, (short)0, acc[nt], false, false);
        }
      }
    }
    // C/D layout: vgpr v -> row v (lanes 0-15) / v+8 (lanes 16-31), col = lane&15
    #pragma unroll
    for (int nt = 0; nt < 8; nt++) {
      #pragma unroll
      for (int v = 0; v < 8; v++) {
        int r = tile * 16 + v + hb * 8;
        if (r < N) {
          int c = nt * 16 + row15;
          float val = acc[nt][v];
          if (m == 0)      Qh[(size_t)r * HID + c] = (_Float16)val;
          else if (m == 1) Kh[(size_t)r * HID + c] = (_Float16)val;
          else             Vf[(size_t)r * HID + c] = val;
        }
      }
    }
  }
}

// ---------- phase 2: Kt[n, et, :] = K[n] @ W_edge[et]  (dense WMMA) ----------

__global__ void __launch_bounds__(256) k_edge_tf(const _Float16* Kh,
                                                 const _Float16* Wf,
                                                 _Float16* Kt, int N) {
  int nTiles = (N + 15) >> 4;
  int widx = blockIdx.x * 8 + (threadIdx.x >> 5);
  int tile = widx >> 3;
  int et = widx & 7;
  if (tile >= nTiles) return;                  // wave-uniform
  int lane = threadIdx.x & 31;
  int row15 = lane & 15;
  int hb = lane >> 4;
  int node = tile * 16 + row15;
  int nodeC = node < N ? node : N - 1;

  f16x16 a[4];
  #pragma unroll
  for (int kc = 0; kc < 4; kc++) {
    const _Float16* p = Kh + (size_t)nodeC * HID + kc * 32 + hb * 8;
    f16x8 lo = *(const f16x8*)p;
    f16x8 hi = *(const f16x8*)(p + 16);
    #pragma unroll
    for (int j = 0; j < 8; j++) { a[kc][j] = lo[j]; a[kc][j + 8] = hi[j]; }
  }

  #pragma unroll
  for (int nt = 0; nt < 8; nt++) {
    f32x8 acc;
    #pragma unroll
    for (int v = 0; v < 8; v++) acc[v] = 0.f;
    #pragma unroll
    for (int kc = 0; kc < 4; kc++) {
      f16x16 b = load_bfrag(Wf, 3 * NTY + et, kc, nt, lane);
      acc = __builtin_amdgcn_wmma_f32_16x16x32_f16(
          false, a[kc], false, b, (short)0, acc, false, false);
    }
    #pragma unroll
    for (int v = 0; v < 8; v++) {
      int r = tile * 16 + v + hb * 8;
      if (r < N)
        Kt[((size_t)r * ETY + et) * HID + nt * 16 + row15] = (_Float16)acc[v];
    }
  }
}

// ---------- phase 3a: scores + segment max ----------
// 8 lanes per edge; each lane loads 32B of Qh[dst] and Kt[src,et] (L2 resident).

__global__ void __launch_bounds__(256) k_scores(const _Float16* Qh,
                                                const _Float16* Kt,
                                                const int* src, const int* dst,
                                                const int* etype, const float* mu,
                                                float* scores, unsigned* segmax,
                                                int E) {
  int e = blockIdx.x * 32 + (threadIdx.x >> 3);
  if (e >= E) return;
  int l = threadIdx.x & 7;
  int s = src[e], d = dst[e], et = etype[e];
  f16x16 q = *(const f16x16*)(Qh + (size_t)d * HID + l * 16);
  f16x16 k = *(const f16x16*)(Kt + ((size_t)s * ETY + et) * HID + l * 16);
  float acc = 0.f;
  #pragma unroll
  for (int j = 0; j < 16; j++) acc += (float)q[j] * (float)k[j];
  acc += __shfl_xor(acc, 1);
  acc += __shfl_xor(acc, 2);
  acc += __shfl_xor(acc, 4);
  if (l == 0) {
    float sc = acc * mu[et] * 0.08838834764831845f;  // 1/sqrt(128)
    scores[e] = sc;
    atomicMax(segmax + d, encF(sc));
  }
}

// ---------- phase 3b: exp + segment sum ----------

__global__ void __launch_bounds__(256) k_norm(float* scores, const int* dst,
                                              const unsigned* segmax,
                                              float* segsum, int E) {
  int e = blockIdx.x * 256 + threadIdx.x;
  if (e >= E) return;
  int d = dst[e];
  float ex = expf(scores[e] - decF(segmax[d]));
  scores[e] = ex;
  atomicAdd(segsum + d, ex);
}

// ---------- phase 3c: out[dst] += attn * V[src] ----------
// One wave per edge, 4 columns per lane.

__global__ void __launch_bounds__(256) k_agg(const float* scores,
                                             const float* segsum,
                                             const int* src, const int* dst,
                                             const float* Vf, float* out, int E) {
  int e = blockIdx.x * 8 + (threadIdx.x >> 5);
  if (e >= E) return;
  int lane = threadIdx.x & 31;
  int s = src[e], d = dst[e];
  float attn = scores[e] / (segsum[d] + 1e-10f);
  #pragma unroll
  for (int j = 0; j < 4; j++) {
    int c = lane + 32 * j;
    atomicAdd(out + (size_t)d * HID + c, attn * Vf[(size_t)s * HID + c]);
  }
}

// ---------- launcher ----------

extern "C" void kernel_launch(void* const* d_in, const int* in_sizes, int n_in,
                              void* d_out, int out_size, void* d_ws, size_t ws_size,
                              hipStream_t stream) {
  (void)n_in; (void)out_size; (void)ws_size;
  const float* x   = (const float*)d_in[0];
  const int* eidx  = (const int*)d_in[1];   // int32 (jax x64 off)
  const int* etype = (const int*)d_in[2];
  const int* ntype = (const int*)d_in[3];
  const float* WQ  = (const float*)d_in[4];
  const float* WK  = (const float*)d_in[5];
  const float* WV  = (const float*)d_in[6];
  const float* WE  = (const float*)d_in[7];
  const float* mu  = (const float*)d_in[8];
  const int N = in_sizes[0] / HID;
  const int E = in_sizes[2];
  const int* src = eidx;
  const int* dst = eidx + E;
  float* out = (float*)d_out;

  // Workspace carve (~158 MB): Qh/Kh f16, V f32, Kt f16, scores, seg bufs, W frags.
  char* w = (char*)d_ws;
  size_t off = 0;
  auto alloc = [&](size_t bytes) -> char* {
    char* p = w + off;
    off = (off + bytes + 255) & ~(size_t)255;
    return p;
  };
  _Float16* Qh     = (_Float16*)alloc((size_t)N * HID * 2);
  _Float16* Kh     = (_Float16*)alloc((size_t)N * HID * 2);
  float*    Vf     = (float*)   alloc((size_t)N * HID * 4);
  _Float16* Kt     = (_Float16*)alloc((size_t)N * ETY * HID * 2);
  float*    scores = (float*)   alloc((size_t)E * 4);
  unsigned* segmax = (unsigned*)alloc((size_t)N * 4);
  float*    segsum = (float*)   alloc((size_t)N * 4);
  _Float16* Wf     = (_Float16*)alloc((size_t)(3 * NTY + ETY) * HID * HID * 2);

  int nTiles = (N + 15) / 16;
  k_init<<<(N * HID + 255) / 256, 256, 0, stream>>>(out, segmax, segsum, N);
  k_pack_w<<<((3 * NTY + ETY) * HID * HID + 255) / 256, 256, 0, stream>>>(WQ, WK, WV, WE, Wf);
  k_project<<<(nTiles + 7) / 8, 256, 0, stream>>>(x, ntype, Wf, Qh, Kh, Vf, N);
  k_edge_tf<<<(nTiles * ETY + 7) / 8, 256, 0, stream>>>(Kh, Wf, Kt, N);
  k_scores<<<(E + 31) / 32, 256, 0, stream>>>(Qh, Kt, src, dst, etype, mu, scores, segmax, E);
  k_norm<<<(E + 255) / 256, 256, 0, stream>>>(scores, dst, segmax, segsum, E);
  k_agg<<<(E + 7) / 8, 256, 0, stream>>>(scores, segsum, src, dst, Vf, out, E);
}
